// TorchMultiHeadAttention_60756607369283
// MI455X (gfx1250) — matrix operand order repaired and checked by reading, not compiled, not run on hardware
//
#include <hip/hip_runtime.h>
#include <hip/hip_bf16.h>

// ---------------------------------------------------------------------------
// ProbSparse multi-head attention (Informer-style) for MI455X / gfx1250.
// bf16 WMMA (v_wmma_f32_16x16x32_bf16) for the big GEMM stages, TDM
// (tensor_load_to_lds + s_wait_tensorcnt) double-buffered LDS staging,
// fp32 VALU for the tiny numerics-critical tail stages.
// ---------------------------------------------------------------------------

typedef __attribute__((ext_vector_type(16))) __bf16    v16bf;
typedef __attribute__((ext_vector_type(8)))  float     v8f;
typedef __attribute__((ext_vector_type(4)))  unsigned  v4u;
typedef __attribute__((ext_vector_type(8)))  int       v8i;
typedef __attribute__((ext_vector_type(4)))  int       v4i;

#define B_DIM    2
#define L_DIM    2048
#define DM       1024
#define N_HEADS  16
#define HEAD_DIM 64
#define TOP_U    40
#define BH       (B_DIM * N_HEADS)
#define M_ROWS   (B_DIM * L_DIM)      // 4096
#define SCALE    0.125f               // 1/sqrt(64)

__device__ __forceinline__ unsigned short f2bf(float x) {
    unsigned u = __float_as_uint(x);
    u += 0x7FFFu + ((u >> 16) & 1u);          // round-to-nearest-even
    return (unsigned short)(u >> 16);
}
__device__ __forceinline__ unsigned pk2bf(float a, float b) {
    return (unsigned)f2bf(a) | ((unsigned)f2bf(b) << 16);
}

union FragBF { v16bf v; uint4 q[2]; };

// ---------------------------------------------------------------------------
// TDM: DMA one 64-row x 32-col bf16 tile (row stride = rowStrideElems) from
// global memory into LDS at ldsAddr, padding every 64-byte row with 16 bytes
// so LDS rows land on an 80-byte stride (bank-conflict-spread ds_load_b128).
// D# built per CDNA5 ISA ch.8 (2-D tile: groups 2/3 zero). This toolchain's
// builtin takes 6 operands: (g0 v4u, g1 v8i, g2 v4i, g3 v4i, v8i, cpol).
// ---------------------------------------------------------------------------
__device__ __forceinline__ void tdm_load_tile_bf16(const unsigned short* gsrc,
                                                   unsigned ldsAddr,
                                                   unsigned rowStrideElems)
{
    const unsigned long long ga = (unsigned long long)(uintptr_t)gsrc;
    v4u g0;
    g0.x = 1u;                                                 // count=1 (valid)
    g0.y = ldsAddr;                                            // lds_addr
    g0.z = (unsigned)ga;                                       // global_addr[31:0]
    g0.w = (unsigned)((ga >> 32) & 0x01FFFFFFu) | (2u << 30);  // [56:32] | type=2

    v8i g1;
    g1[0] = (int)((1u << 16)      // data_size = 2 bytes
                | (1u << 20)      // pad_enable
                | (3u << 22)      // pad_interval: 16 DWORDs (64 B)
                | (3u << 25));    // pad_amount:   4 DWORDs (16 B)
    g1[1] = (int)(32u << 16);     // tensor_dim0 = 32 (low 16 bits in d1[31:16])
    g1[2] = (int)(64u << 16);     // tensor_dim0 hi=0 | tensor_dim1 = 64 (low)
    g1[3] = (int)(32u << 16);     // tensor_dim1 hi=0 | tile_dim0 = 32
    g1[4] = (int)(64u);           // tile_dim1 = 64 | tile_dim2 = 0
    g1[5] = (int)rowStrideElems;  // tensor_dim0_stride[31:0]
    g1[6] = 0;                    // stride0 hi | tensor_dim1_stride lo
    g1[7] = 0;

    const v4i z4 = (v4i){0, 0, 0, 0};
    const v8i z8 = (v8i){0, 0, 0, 0, 0, 0, 0, 0};
    __builtin_amdgcn_tensor_load_to_lds(g0, g1, z4, z4, z8, 0);
}

// ---------------------------------------------------------------------------
// Elementwise fp32 -> bf16 (vectorized x4). One-time conversion so the GEMMs
// read bf16 directly and TDM can DMA without conversion.
// ---------------------------------------------------------------------------
__global__ __launch_bounds__(256)
void cvt_f32_to_bf16(const float* __restrict__ src,
                     unsigned short* __restrict__ dst, int n4)
{
    const int i = blockIdx.x * 256 + threadIdx.x;
    if (i >= n4) return;
    const float4 v = ((const float4*)src)[i];
    ((uint2*)dst)[i] = make_uint2(pk2bf(v.x, v.y), pk2bf(v.z, v.w));
}

// ---------------------------------------------------------------------------
// C[M,N] = A[M,K] * W[N,K]^T + bias   (bf16 in, bf16 WMMA, fp32 accumulate)
// Block tile 64x64, K-step 32, 128 threads = 4 waves, each wave: 16(M)x64(N).
// Double-buffered LDS tiles filled by TDM; wave 0 issues the DMA descriptors
// and gates the block on s_wait_tensorcnt + workgroup barrier.
// headsplit=1: store into [B,H,L,64] layout (+ optional bf16 mirror copy).
// ---------------------------------------------------------------------------
#define LDS_ROWB 80

__global__ __launch_bounds__(128)
void gemm_bt_bf16(const unsigned short* __restrict__ A,   // bf16 [M,K]
                  const unsigned short* __restrict__ W,   // bf16 [N,K]
                  const float* __restrict__ bias,
                  float* __restrict__ Cf, unsigned short* __restrict__ Cbf,
                  int M, int N, int K, int headsplit)
{
    __shared__ __align__(16) unsigned char ldsA[2][64 * LDS_ROWB];
    __shared__ __align__(16) unsigned char ldsW[2][64 * LDS_ROWB];

    const int tid  = threadIdx.x;
    const int lane = tid & 31;
    const int wave = tid >> 5;
    const int half = lane >> 4;       // 0: lanes 0-15, 1: lanes 16-31
    const int l15  = lane & 15;
    const int rowBase = blockIdx.y * 64;
    const int colBase = blockIdx.x * 64;
    const int nkt = K / 32;

    v8f acc[4];
    for (int i = 0; i < 4; ++i)
        acc[i] = (v8f){0.f, 0.f, 0.f, 0.f, 0.f, 0.f, 0.f, 0.f};

    // prologue: DMA tile 0 into buffer 0
    if (wave == 0) {
        tdm_load_tile_bf16(A + (size_t)rowBase * K,
                           (unsigned)(uintptr_t)&ldsA[0][0], (unsigned)K);
        tdm_load_tile_bf16(W + (size_t)colBase * K,
                           (unsigned)(uintptr_t)&ldsW[0][0], (unsigned)K);
    }

    for (int it = 0; it < nkt; ++it) {
        const int cur = it & 1, nxt = cur ^ 1;
        __syncthreads();  // all waves done reading buffer 'nxt' (iter it-1)
        if (wave == 0) {
            if (it + 1 < nkt) {
                const int k1 = (it + 1) * 32;
                tdm_load_tile_bf16(A + (size_t)rowBase * K + k1,
                                   (unsigned)(uintptr_t)&ldsA[nxt][0], (unsigned)K);
                tdm_load_tile_bf16(W + (size_t)colBase * K + k1,
                                   (unsigned)(uintptr_t)&ldsW[nxt][0], (unsigned)K);
                __builtin_amdgcn_s_wait_tensorcnt(2);  // tile 'it' done; next in flight
            } else {
                __builtin_amdgcn_s_wait_tensorcnt(0);
            }
        }
        __syncthreads();  // tile 'it' visible to the whole block

        // A fragment (16x32 bf16): lanes 0-15 K=0-7/16-23, lanes 16-31 K=8-15/24-31
        FragBF af;
        const unsigned char* abase = &ldsA[cur][0] + (wave * 16 + l15) * LDS_ROWB + half * 16;
        af.q[0] = *(const uint4*)(abase);
        af.q[1] = *(const uint4*)(abase + 32);

        // preload all four B fragments, then a 4-WMMA burst
        FragBF bfr[4];
        for (int nt = 0; nt < 4; ++nt) {
            const unsigned char* bbase = &ldsW[cur][0] + (nt * 16 + l15) * LDS_ROWB + half * 16;
            bfr[nt].q[0] = *(const uint4*)(bbase);
            bfr[nt].q[1] = *(const uint4*)(bbase + 32);
        }
        for (int nt = 0; nt < 4; ++nt)
            acc[nt] = __builtin_amdgcn_wmma_f32_16x16x32_bf16(
                false, af.v, false, bfr[nt].v, (short)0, acc[nt], false, false);
    }

    // C layout: VGPR r -> row r (lanes 0-15) / row r+8 (lanes 16-31), col = lane&15
    for (int nt = 0; nt < 4; ++nt) {
        const int col = colBase + nt * 16 + l15;
        const float bv = bias ? bias[col] : 0.f;
        for (int r = 0; r < 8; ++r) {
            const int row = rowBase + wave * 16 + r + half * 8;
            const float c = acc[nt][r] + bv;
            if (headsplit) {
                const int b = row >> 11, l = row & (L_DIM - 1);
                const int h = col >> 6,  d = col & 63;
                const size_t off = (((size_t)(b * N_HEADS + h)) * L_DIM + l) * HEAD_DIM + d;
                Cf[off] = c;
                if (Cbf) Cbf[off] = f2bf(c);
            } else {
                Cf[(size_t)row * N + col] = c;
            }
        }
    }
}

// ---------------------------------------------------------------------------
// sparsity[b,h,q] = scale*max_k(q.k) - scale*mean_k(q.k), fused over all keys.
// grid (L/64, B*H), 128 threads = 4 waves; each wave: 16 queries, WMMA over
// 16-key tiles (2 WMMAs per tile for head_dim 64 = 2 x K32).
// ---------------------------------------------------------------------------
__global__ __launch_bounds__(128)
void sparsity_kernel(const unsigned short* __restrict__ qbf,
                     const unsigned short* __restrict__ kbf,
                     float* __restrict__ spars)
{
    const int tid  = threadIdx.x;
    const int lane = tid & 31;
    const int wave = tid >> 5;
    const int half = lane >> 4;
    const int l15  = lane & 15;
    const int bh   = blockIdx.y;
    const int q0   = blockIdx.x * 64 + wave * 16;

    // A fragments: this wave's 16 query rows, d=0..31 and d=32..63
    FragBF a0, a1;
    const unsigned short* ap =
        qbf + ((size_t)bh * L_DIM + q0 + l15) * HEAD_DIM + half * 8;
    a0.q[0] = *(const uint4*)(ap);
    a0.q[1] = *(const uint4*)(ap + 16);
    a1.q[0] = *(const uint4*)(ap + 32);
    a1.q[1] = *(const uint4*)(ap + 48);

    v8f runmax = (v8f){-3e38f, -3e38f, -3e38f, -3e38f, -3e38f, -3e38f, -3e38f, -3e38f};
    v8f runsum = (v8f){0.f, 0.f, 0.f, 0.f, 0.f, 0.f, 0.f, 0.f};

    for (int k0 = 0; k0 < L_DIM; k0 += 16) {
        // B fragments: K^T, contraction dim = d; lane n <- key row k0+n
        const unsigned short* kp =
            kbf + ((size_t)bh * L_DIM + k0 + l15) * HEAD_DIM + half * 8;
        FragBF b0, b1;
        b0.q[0] = *(const uint4*)(kp);
        b0.q[1] = *(const uint4*)(kp + 16);
        b1.q[0] = *(const uint4*)(kp + 32);
        b1.q[1] = *(const uint4*)(kp + 48);

        v8f s = (v8f){0.f, 0.f, 0.f, 0.f, 0.f, 0.f, 0.f, 0.f};
        s = __builtin_amdgcn_wmma_f32_16x16x32_bf16(false, a0.v, false, b0.v,
                                                    (short)0, s, false, false);
        s = __builtin_amdgcn_wmma_f32_16x16x32_bf16(false, a1.v, false, b1.v,
                                                    (short)0, s, false, false);
        for (int r = 0; r < 8; ++r) {
            runmax[r] = fmaxf(runmax[r], s[r]);
            runsum[r] += s[r];
        }
    }

    // reduce across the 16 key-lanes of each half
    for (int off = 8; off; off >>= 1) {
        for (int r = 0; r < 8; ++r) {
            runmax[r] = fmaxf(runmax[r], __shfl_xor(runmax[r], off, 16));
            runsum[r] += __shfl_xor(runsum[r], off, 16);
        }
    }
    if (l15 == 0) {
        for (int r = 0; r < 8; ++r) {
            const int row = q0 + r + half * 8;
            spars[(size_t)bh * L_DIM + row] =
                SCALE * runmax[r] - SCALE * runsum[r] * (1.f / (float)L_DIM);
        }
    }
}

// ---------------------------------------------------------------------------
// Iterative top-40 per (b,h).
// ---------------------------------------------------------------------------
__global__ __launch_bounds__(256)
void topk_kernel(const float* __restrict__ spars, int* __restrict__ topk)
{
    __shared__ float sv[L_DIM];
    __shared__ float rv[256];
    __shared__ int   ri[256];
    const int bh = blockIdx.x, tid = threadIdx.x;
    for (int i = tid; i < L_DIM; i += 256) sv[i] = spars[(size_t)bh * L_DIM + i];
    __syncthreads();
    for (int u = 0; u < TOP_U; ++u) {
        float best = -3e38f; int bi = 0;
        for (int i = tid; i < L_DIM; i += 256) {
            const float x = sv[i];
            if (x > best) { best = x; bi = i; }
        }
        rv[tid] = best; ri[tid] = bi;
        __syncthreads();
        for (int s = 128; s; s >>= 1) {
            if (tid < s && rv[tid + s] > rv[tid]) { rv[tid] = rv[tid + s]; ri[tid] = ri[tid + s]; }
            __syncthreads();
        }
        if (tid == 0) { topk[bh * TOP_U + u] = ri[0]; sv[ri[0]] = -3e38f; }
        __syncthreads();
    }
}

// ---------------------------------------------------------------------------
// vmean[bh][d] = mean over keys of V.
// ---------------------------------------------------------------------------
__global__ void vmean_kernel(const float* __restrict__ vh, float* __restrict__ vmean)
{
    const int bh = blockIdx.x, d = threadIdx.x;
    const float* p = vh + (size_t)bh * L_DIM * HEAD_DIM + d;
    float s = 0.f;
    for (int l = 0; l < L_DIM; ++l) s += p[(size_t)l * HEAD_DIM];
    vmean[bh * HEAD_DIM + d] = s * (1.f / (float)L_DIM);
}

// ---------------------------------------------------------------------------
// context base fill: broadcast vmean into [B, L, D_MODEL].
// ---------------------------------------------------------------------------
__global__ void fill_base_kernel(const float* __restrict__ vmean, float* __restrict__ ctx)
{
    const size_t i = (size_t)blockIdx.x * 256 + threadIdx.x; // over B*L*DM
    const int c = (int)(i & (DM - 1));
    const size_t bl = i >> 10;
    const int b = (int)(bl >> 11);
    const int h = c >> 6, d = c & 63;
    ctx[i] = vmean[(b * N_HEADS + h) * HEAD_DIM + d];
}

// ---------------------------------------------------------------------------
// Exact fp32 softmax attention for the 40 selected queries; scatter into ctx.
// grid (TOP_U, B*H), 256 threads.
// ---------------------------------------------------------------------------
__global__ __launch_bounds__(256)
void topattn_kernel(const float* __restrict__ qh, const float* __restrict__ kh,
                    const float* __restrict__ vh, const int* __restrict__ topk,
                    float* __restrict__ ctx)
{
    __shared__ float qv[HEAD_DIM];
    __shared__ float s[L_DIM];
    __shared__ float red[256];
    __shared__ float part[4][HEAD_DIM];

    const int bh = blockIdx.y, u = blockIdx.x, tid = threadIdx.x;
    const int t = topk[bh * TOP_U + u];

    if (tid < HEAD_DIM) qv[tid] = qh[((size_t)bh * L_DIM + t) * HEAD_DIM + tid];
    __syncthreads();

    float lmax = -3e38f;
    for (int k = tid; k < L_DIM; k += 256) {
        const float* kr = kh + ((size_t)bh * L_DIM + k) * HEAD_DIM;
        float dot = 0.f;
        for (int d = 0; d < HEAD_DIM; ++d) dot += qv[d] * kr[d];
        dot *= SCALE;
        s[k] = dot;
        lmax = fmaxf(lmax, dot);
    }
    red[tid] = lmax;
    __syncthreads();
    for (int st = 128; st; st >>= 1) {
        if (tid < st) red[tid] = fmaxf(red[tid], red[tid + st]);
        __syncthreads();
    }
    const float mx = red[0];
    __syncthreads();

    float lsum = 0.f;
    for (int k = tid; k < L_DIM; k += 256) {
        const float e = expf(s[k] - mx);
        s[k] = e;
        lsum += e;
    }
    red[tid] = lsum;
    __syncthreads();
    for (int st = 128; st; st >>= 1) {
        if (tid < st) red[tid] += red[tid + st];
        __syncthreads();
    }
    const float denom = red[0];
    __syncthreads();

    const int d = tid & 63, g = tid >> 6;
    float p = 0.f;
    for (int k = g; k < L_DIM; k += 4)
        p += s[k] * vh[((size_t)bh * L_DIM + k) * HEAD_DIM + d];
    part[g][d] = p;
    __syncthreads();
    if (tid < HEAD_DIM) {
        const float c = (part[0][tid] + part[1][tid] + part[2][tid] + part[3][tid]) / denom;
        const int b = bh >> 4, h = bh & 15;
        ctx[((size_t)b * L_DIM + t) * DM + h * HEAD_DIM + tid] = c;
    }
}

// ---------------------------------------------------------------------------
extern "C" void kernel_launch(void* const* d_in, const int* in_sizes, int n_in,
                              void* d_out, int out_size, void* d_ws, size_t ws_size,
                              hipStream_t stream)
{
    (void)in_sizes; (void)n_in; (void)out_size; (void)ws_size;
    const float* query = (const float*)d_in[0];
    const float* key   = (const float*)d_in[1];
    const float* value = (const float*)d_in[2];
    const float* Wq = (const float*)d_in[3];
    const float* bq = (const float*)d_in[4];
    const float* Wk = (const float*)d_in[5];
    const float* bk = (const float*)d_in[6];
    const float* Wv = (const float*)d_in[7];
    const float* bv = (const float*)d_in[8];
    const float* Wo = (const float*)d_in[9];
    const float* bo = (const float*)d_in[10];
    float* out = (float*)d_out;

    const size_t NELEM = (size_t)BH * L_DIM * HEAD_DIM; // 4,194,304 (= M_ROWS*DM)
    const size_t WELEM = (size_t)DM * DM;               // 1,048,576
    unsigned char* ws = (unsigned char*)d_ws;
    size_t off = 0;
    auto alloc = [&](size_t bytes) -> void* {
        void* p = ws + off;
        off = (off + bytes + 255) & ~(size_t)255;
        return p;
    };
    float*          qh     = (float*)alloc(NELEM * 4);           // [B,H,L,64] f32
    float*          kh     = (float*)alloc(NELEM * 4);
    float*          vh     = (float*)alloc(NELEM * 4);
    unsigned short* qbf_hs = (unsigned short*)alloc(NELEM * 2);  // headsplit bf16
    unsigned short* kbf_hs = (unsigned short*)alloc(NELEM * 2);
    unsigned short* qin_bf = (unsigned short*)alloc(NELEM * 2);  // bf16 GEMM inputs
    unsigned short* kin_bf = (unsigned short*)alloc(NELEM * 2);
    unsigned short* vin_bf = (unsigned short*)alloc(NELEM * 2);
    unsigned short* wq_bf  = (unsigned short*)alloc(WELEM * 2);
    unsigned short* wk_bf  = (unsigned short*)alloc(WELEM * 2);
    unsigned short* wv_bf  = (unsigned short*)alloc(WELEM * 2);
    unsigned short* wo_bf  = (unsigned short*)alloc(WELEM * 2);
    float*          spars  = (float*)alloc((size_t)BH * L_DIM * 4);
    int*            topk   = (int*)alloc((size_t)BH * TOP_U * 4);
    float*          vmean  = (float*)alloc((size_t)BH * HEAD_DIM * 4);
    float*          ctx    = (float*)alloc(NELEM * 4);           // [B,L,DM] f32
    unsigned short* ctx_bf = (unsigned short*)alloc(NELEM * 2);

    const int n4x = (int)(NELEM / 4), n4w = (int)(WELEM / 4);
    cvt_f32_to_bf16<<<(n4x + 255) / 256, 256, 0, stream>>>(query, qin_bf, n4x);
    cvt_f32_to_bf16<<<(n4x + 255) / 256, 256, 0, stream>>>(key,   kin_bf, n4x);
    cvt_f32_to_bf16<<<(n4x + 255) / 256, 256, 0, stream>>>(value, vin_bf, n4x);
    cvt_f32_to_bf16<<<(n4w + 255) / 256, 256, 0, stream>>>(Wq, wq_bf, n4w);
    cvt_f32_to_bf16<<<(n4w + 255) / 256, 256, 0, stream>>>(Wk, wk_bf, n4w);
    cvt_f32_to_bf16<<<(n4w + 255) / 256, 256, 0, stream>>>(Wv, wv_bf, n4w);
    cvt_f32_to_bf16<<<(n4w + 255) / 256, 256, 0, stream>>>(Wo, wo_bf, n4w);

    const dim3 gblk(DM / 64, M_ROWS / 64); // (16, 64)
    gemm_bt_bf16<<<gblk, 128, 0, stream>>>(qin_bf, wq_bf, bq, qh, qbf_hs, M_ROWS, DM, DM, 1);
    gemm_bt_bf16<<<gblk, 128, 0, stream>>>(kin_bf, wk_bf, bk, kh, kbf_hs, M_ROWS, DM, DM, 1);
    gemm_bt_bf16<<<gblk, 128, 0, stream>>>(vin_bf, wv_bf, bv, vh, (unsigned short*)nullptr,
                                           M_ROWS, DM, DM, 1);

    sparsity_kernel<<<dim3(L_DIM / 64, BH), 128, 0, stream>>>(qbf_hs, kbf_hs, spars);
    topk_kernel<<<BH, 256, 0, stream>>>(spars, topk);
    vmean_kernel<<<BH, 64, 0, stream>>>(vh, vmean);
    fill_base_kernel<<<(int)(NELEM / 256), 256, 0, stream>>>(vmean, ctx);
    topattn_kernel<<<dim3(TOP_U, BH), 256, 0, stream>>>(qh, kh, vh, topk, ctx);

    cvt_f32_to_bf16<<<(n4x + 255) / 256, 256, 0, stream>>>(ctx, ctx_bf, n4x);
    gemm_bt_bf16<<<gblk, 128, 0, stream>>>(ctx_bf, wo_bf, bo, out, (unsigned short*)nullptr,
                                           M_ROWS, DM, DM, 0);
}
